// ScaledSelfAttention_25228637897060
// MI455X (gfx1250) — compile-verified, matrix operand order checked by reading, not run on hardware
//
#include <hip/hip_runtime.h>
#include <hip/hip_bf16.h>

#define EMB  256
#define QD   512
#define SEQ  2048
#define NBAT 8

typedef __attribute__((ext_vector_type(16))) __bf16 v16bf;
typedef __attribute__((ext_vector_type(8)))  __bf16 v8bf;
typedef __attribute__((ext_vector_type(8)))  float  v8f;

static __device__ __forceinline__ unsigned short f2bf(float f) {
  union { float f; unsigned u; } v; v.f = f;
  unsigned r = v.u + 0x7FFFu + ((v.u >> 16) & 1u);   // round-to-nearest-even
  return (unsigned short)(r >> 16);
}
static __device__ __forceinline__ v8bf ld8(const unsigned short* p) {
  return *reinterpret_cast<const v8bf*>(p);
}
static __device__ __forceinline__ v16bf cat16(v8bf a, v8bf b) {
  return __builtin_shufflevector(a, b, 0,1,2,3,4,5,6,7,8,9,10,11,12,13,14,15);
}

// gfx1250 async global->LDS copy, 16B per lane, tracked by ASYNCcnt.
// GV mode: lds[vdst + ioffset] = mem[vaddr64 + ioffset]
static __device__ __forceinline__ void async_ld16(unsigned lds_off,
                                                  const void* gaddr) {
  asm volatile("global_load_async_to_lds_b128 %0, %1, off"
               :: "v"(lds_off), "v"(gaddr) : "memory");
}
static __device__ __forceinline__ void wait_async0() {
  asm volatile("s_wait_asynccnt 0x0" ::: "memory");
}

// ---------------------------------------------------------------------------
// Kernel 1: convert W[e][o] (f32, [256,512]) -> Wt[o][e] (bf16, [512,256])
// ---------------------------------------------------------------------------
__global__ void prep_weights(const float* __restrict__ Wq,
                             const float* __restrict__ Wk,
                             const float* __restrict__ Wv,
                             unsigned short* __restrict__ Wqt,
                             unsigned short* __restrict__ Wkt,
                             unsigned short* __restrict__ Wvt) {
  int idx = blockIdx.x * blockDim.x + threadIdx.x;   // 3 * 512 * 256
  int mat = idx >> 17;
  int r   = idx & 131071;
  int o   = r >> 8;      // 0..511
  int e   = r & 255;     // 0..255
  const float* W = (mat == 0) ? Wq : (mat == 1) ? Wk : Wv;
  unsigned short* Wt = (mat == 0) ? Wqt : (mat == 1) ? Wkt : Wvt;
  Wt[o * EMB + e] = f2bf(W[e * QD + o]);
}

// ---------------------------------------------------------------------------
// Kernel 2: projections. q = sigmoid(xWq+bq) -> bf16 [16384,512]
//                        k = xWk+bk         -> bf16 [16384,512]
//                        vT = (xWv+bv)^T    -> bf16 [8,512,2048]
// ---------------------------------------------------------------------------
__global__ __launch_bounds__(256) void proj_kernel(
    const float* __restrict__ x,
    const unsigned short* __restrict__ Wqt,
    const unsigned short* __restrict__ Wkt,
    const unsigned short* __restrict__ Wvt,
    const float* __restrict__ bq, const float* __restrict__ bk,
    const float* __restrict__ bv,
    unsigned short* __restrict__ qb, unsigned short* __restrict__ kb,
    unsigned short* __restrict__ vtb) {
  const int lane = threadIdx.x & 31;
  const int wave = threadIdx.x >> 5;
  const int h    = lane >> 4;
  const int lm   = lane & 15;
  const int rb   = blockIdx.x * 128 + wave * 16;   // global row base
  const int mat  = blockIdx.y >> 2;
  const int n0   = (blockIdx.y & 3) * 8;

  // A fragments: rows rb..rb+15, features 0..255, converted f32->bf16
  v16bf A[8];
#pragma unroll
  for (int f = 0; f < 8; ++f) {
    const float* px = x + (size_t)(rb + lm) * EMB + f * 32 + h * 8;
    const float4* p0 = reinterpret_cast<const float4*>(px);
    const float4* p1 = reinterpret_cast<const float4*>(px + 16);
    float4 a0 = p0[0], a1 = p0[1], b0 = p1[0], b1 = p1[1];
    unsigned short tmp[16] = {
      f2bf(a0.x), f2bf(a0.y), f2bf(a0.z), f2bf(a0.w),
      f2bf(a1.x), f2bf(a1.y), f2bf(a1.z), f2bf(a1.w),
      f2bf(b0.x), f2bf(b0.y), f2bf(b0.z), f2bf(b0.w),
      f2bf(b1.x), f2bf(b1.y), f2bf(b1.z), f2bf(b1.w)};
    v16bf av; __builtin_memcpy(&av, tmp, 32); A[f] = av;
  }

  const unsigned short* Wt = (mat == 0) ? Wqt : (mat == 1) ? Wkt : Wvt;
  const float* bias        = (mat == 0) ? bq  : (mat == 1) ? bk  : bv;

  for (int nt = 0; nt < 8; ++nt) {
    const int col = (n0 + nt) * 16 + lm;
    v8f acc = {};
#pragma unroll
    for (int f = 0; f < 8; ++f) {
      const unsigned short* pw = Wt + (size_t)col * EMB + f * 32 + h * 8;
      v16bf B = cat16(ld8(pw), ld8(pw + 16));
      acc = __builtin_amdgcn_wmma_f32_16x16x32_bf16(
          false, A[f], false, B, (short)0, acc, false, false);
    }
    float bvv = bias[col];
#pragma unroll
    for (int r = 0; r < 8; ++r) {
      int grow = rb + r + 8 * h;
      float val = acc[r] + bvv;
      if (mat == 0) {
        val = 1.0f / (1.0f + __expf(-val));           // sigmoid
        qb[(size_t)grow * QD + col] = f2bf(val);
      } else if (mat == 1) {
        kb[(size_t)grow * QD + col] = f2bf(val);
      } else {
        int bb = grow >> 11, s = grow & 2047;
        vtb[((size_t)bb * QD + col) * SEQ + s] = f2bf(val);
      }
    }
  }
}

// ---------------------------------------------------------------------------
// Kernel 3: fused flash attention, double-buffered async LDS staging.
// grid = (B * S/64, 2); block = 128 thr = 4 waves.
// ---------------------------------------------------------------------------
#define KS_PITCH 520   // ushorts per K row (1040B: 16B-aligned, 4-bank stride)
#define VS_PITCH 40    // ushorts per V^T row (80B)
#define PS_PITCH 40
#define KS_ELEMS (32 * KS_PITCH)
#define VS_ELEMS (256 * VS_PITCH)
#define SMEM_USHORTS (2 * KS_ELEMS + 2 * VS_ELEMS + 4 * 16 * PS_PITCH)

// issue one tile's worth of async global->LDS copies (this thread's share)
static __device__ __forceinline__ void stage_tile_async(
    unsigned ksBase, unsigned vsBase,
    const unsigned short* __restrict__ kSrc,
    const unsigned short* __restrict__ vSrc, int t) {
  // K tile: 32 rows x 512 bf16 = 2048 x 16B chunks
  for (int c = t; c < 2048; c += 128) {
    int row = c >> 6, off = (c & 63) * 8;
    async_ld16(ksBase + (unsigned)(row * KS_PITCH + off) * 2,
               kSrc + row * QD + off);
  }
  // V^T tile: 256 cols x 32 keys = 1024 x 16B chunks
  for (int c = t; c < 1024; c += 128) {
    int col = c >> 2, off = (c & 3) * 8;
    async_ld16(vsBase + (unsigned)(col * VS_PITCH + off) * 2,
               vSrc + (size_t)col * SEQ + off);
  }
}

__global__ __launch_bounds__(128) void attn_kernel(
    const unsigned short* __restrict__ qb,
    const unsigned short* __restrict__ kb,
    const unsigned short* __restrict__ vtb,
    float* __restrict__ out) {
  extern __shared__ unsigned short smem[];
  unsigned short* Ks0 = smem;
  unsigned short* Ks1 = smem + KS_ELEMS;
  unsigned short* Vs0 = smem + 2 * KS_ELEMS;
  unsigned short* Vs1 = smem + 2 * KS_ELEMS + VS_ELEMS;
  unsigned short* Ps  = smem + 2 * KS_ELEMS + 2 * VS_ELEMS;

  const int t    = threadIdx.x;
  const int lane = t & 31, wave = t >> 5;
  const int h    = lane >> 4, lm = lane & 15;
  const int vh   = blockIdx.y;                       // v-column half
  const int b    = blockIdx.x >> 5;
  const int q0   = (blockIdx.x & 31) * 64 + wave * 16;
  const float scale = 0.0625f;                       // 1/sqrt(256)

  const unsigned short* kbB = kb + (size_t)b * SEQ * QD;
  const unsigned short* vtB = vtb + ((size_t)b * QD + vh * 256) * SEQ;

  // Q fragments resident in VGPRs: 16 frags covering features 0..511
  v16bf Qf[16];
#pragma unroll
  for (int f = 0; f < 16; ++f) {
    const unsigned short* pq =
        qb + ((size_t)b * SEQ + q0 + lm) * QD + f * 32 + h * 8;
    Qf[f] = cat16(ld8(pq), ld8(pq + 16));
  }

  v8f O[16];
#pragma unroll
  for (int j = 0; j < 16; ++j) O[j] = (v8f){};
  float mrow[8], lrow[8];
#pragma unroll
  for (int r = 0; r < 8; ++r) { mrow[r] = -3.0e38f; lrow[r] = 0.f; }

  unsigned short* Pw = Ps + wave * 16 * PS_PITCH;
  const unsigned ks0a = (unsigned)(uintptr_t)Ks0;
  const unsigned ks1a = (unsigned)(uintptr_t)Ks1;
  const unsigned vs0a = (unsigned)(uintptr_t)Vs0;
  const unsigned vs1a = (unsigned)(uintptr_t)Vs1;

  // prologue: stage tile 0 into buffer 0
  stage_tile_async(ks0a, vs0a, kbB, vtB, t);

  for (int kt = 0; kt < SEQ / 32; ++kt) {
    wait_async0();        // this wave's issued copies have landed in LDS
    __syncthreads();      // => whole buffer valid, prev buffer free
    const int cur = kt & 1;
    if (kt + 1 < SEQ / 32) {   // prefetch next tile into the other buffer
      stage_tile_async(cur ? ks0a : ks1a, cur ? vs0a : vs1a,
                       kbB + (size_t)(kt + 1) * 32 * QD, vtB + (kt + 1) * 32, t);
    }
    const unsigned short* Ksc = cur ? Ks1 : Ks0;
    const unsigned short* Vsc = cur ? Vs1 : Vs0;

    // scores S = Q * K^T for two 16-key n-tiles
    v8f S0 = (v8f){}, S1 = (v8f){};
#pragma unroll
    for (int f = 0; f < 16; ++f) {
      const unsigned short* pk = Ksc + lm * KS_PITCH + f * 32 + h * 8;
      v16bf B0 = cat16(ld8(pk), ld8(pk + 16));
      S0 = __builtin_amdgcn_wmma_f32_16x16x32_bf16(
          false, Qf[f], false, B0, (short)0, S0, false, false);
    }
#pragma unroll
    for (int f = 0; f < 16; ++f) {
      const unsigned short* pk = Ksc + (16 + lm) * KS_PITCH + f * 32 + h * 8;
      v16bf B1 = cat16(ld8(pk), ld8(pk + 16));
      S1 = __builtin_amdgcn_wmma_f32_16x16x32_bf16(
          false, Qf[f], false, B1, (short)0, S1, false, false);
    }

    // scale + ELU
#pragma unroll
    for (int r = 0; r < 8; ++r) {
      float a = S0[r] * scale; S0[r] = (a > 0.f) ? a : (__expf(a) - 1.f);
      float c = S1[r] * scale; S1[r] = (c > 0.f) ? c : (__expf(c) - 1.f);
    }

    // online softmax: row max, correction, P, row sum
    float corr[8];
#pragma unroll
    for (int r = 0; r < 8; ++r) {
      float mx = fmaxf(S0[r], S1[r]);
      mx = fmaxf(mx, __shfl_xor(mx, 1, 16));
      mx = fmaxf(mx, __shfl_xor(mx, 2, 16));
      mx = fmaxf(mx, __shfl_xor(mx, 4, 16));
      mx = fmaxf(mx, __shfl_xor(mx, 8, 16));
      float mn = fmaxf(mrow[r], mx);
      corr[r] = __expf(mrow[r] - mn);
      mrow[r] = mn;
    }
#pragma unroll
    for (int r = 0; r < 8; ++r) {
      float p0 = __expf(S0[r] - mrow[r]);
      float p1 = __expf(S1[r] - mrow[r]);
      S0[r] = p0; S1[r] = p1;
      float rs = p0 + p1;
      rs += __shfl_xor(rs, 1, 16);
      rs += __shfl_xor(rs, 2, 16);
      rs += __shfl_xor(rs, 4, 16);
      rs += __shfl_xor(rs, 8, 16);
      lrow[r] = lrow[r] * corr[r] + rs;
    }

    // transpose P (D-layout -> A-layout) through wave-private LDS
#pragma unroll
    for (int r = 0; r < 8; ++r) {
      int m = r + 8 * h;
      Pw[m * PS_PITCH + lm]      = f2bf(S0[r]);
      Pw[m * PS_PITCH + 16 + lm] = f2bf(S1[r]);
    }
    asm volatile("s_wait_dscnt 0" ::: "memory");
    const unsigned short* pp = Pw + lm * PS_PITCH + h * 8;
    v16bf Pf = cat16(ld8(pp), ld8(pp + 16));

    // rescale O and accumulate P*V
#pragma unroll
    for (int j = 0; j < 16; ++j) {
      v8f o = O[j];
#pragma unroll
      for (int r = 0; r < 8; ++r) o[r] *= corr[r];
      const unsigned short* pv = Vsc + (j * 16 + lm) * VS_PITCH + h * 8;
      v16bf Bv = cat16(ld8(pv), ld8(pv + 16));
      O[j] = __builtin_amdgcn_wmma_f32_16x16x32_bf16(
          false, Pf, false, Bv, (short)0, o, false, false);
    }
  }

  // epilogue: divide by l and store f32
#pragma unroll
  for (int j = 0; j < 16; ++j) {
#pragma unroll
    for (int r = 0; r < 8; ++r) {
      int row = q0 + r + 8 * h;
      int col = vh * 256 + j * 16 + lm;
      out[((size_t)b * SEQ + row) * QD + col] = O[j][r] / lrow[r];
    }
  }
}

// ---------------------------------------------------------------------------
extern "C" void kernel_launch(void* const* d_in, const int* in_sizes, int n_in,
                              void* d_out, int out_size, void* d_ws,
                              size_t ws_size, hipStream_t stream) {
  const float* x  = (const float*)d_in[0];
  const float* Wq = (const float*)d_in[1];
  const float* bq = (const float*)d_in[2];
  const float* Wk = (const float*)d_in[3];
  const float* bk = (const float*)d_in[4];
  const float* Wv = (const float*)d_in[5];
  const float* bv = (const float*)d_in[6];
  float* out = (float*)d_out;

  char* ws = (char*)d_ws;
  unsigned short* Wqt = (unsigned short*)ws; ws += (size_t)QD * EMB * 2;
  unsigned short* Wkt = (unsigned short*)ws; ws += (size_t)QD * EMB * 2;
  unsigned short* Wvt = (unsigned short*)ws; ws += (size_t)QD * EMB * 2;
  unsigned short* qb  = (unsigned short*)ws; ws += (size_t)NBAT * SEQ * QD * 2;
  unsigned short* kb  = (unsigned short*)ws; ws += (size_t)NBAT * SEQ * QD * 2;
  unsigned short* vtb = (unsigned short*)ws; ws += (size_t)NBAT * SEQ * QD * 2;

  prep_weights<<<dim3((3 * QD * EMB) / 256), dim3(256), 0, stream>>>(
      Wq, Wk, Wv, Wqt, Wkt, Wvt);
  proj_kernel<<<dim3((NBAT * SEQ) / 128, 12), dim3(256), 0, stream>>>(
      x, Wqt, Wkt, Wvt, bq, bk, bv, qb, kb, vtb);
  attn_kernel<<<dim3((NBAT * SEQ) / 64, 2), dim3(128),
                SMEM_USHORTS * 2, stream>>>(qb, kb, vtb, out);
}
